// Net_67259187855904
// MI455X (gfx1250) — compile-verified, matrix operand order checked by reading, not run on hardware
//
#include <hip/hip_runtime.h>
#include <hip/hip_bf16.h>

#define NB   1024      // N basins
#define LL   1095      // sequence length
#define FF   32        // input features
#define HH   256       // hidden
#define G4   1024      // 4*H
#define KC   512       // concat K = 2*H
#define OUTP 10
#define TMAXUH 15
#define EPSV 0.0003f

typedef __attribute__((ext_vector_type(16))) _Float16 v16h;
typedef __attribute__((ext_vector_type(8)))  float    v8f;

// -------- parameter scaling tables (sigmoid p -> physical ranges) ----------
__device__ __constant__ float c_scale[OUTP] = {-3.f, 5.f, 3.f, 1400.f, 0.1f, 40.f, 1.f, 1.f, 2.9f, 6.5f};
__device__ __constant__ float c_off[OUTP]   = { 0.f, 0.f, 0.f,  100.f, 0.f , 10.f, 0.f, 1.f, 0.f, 0.f};

// Fragment layouts (ISA 7.12.2, wave32, 16-bit A/B):
//   lane l: m/n = l&15, grp = l>>4 ; half j (0..15): K(j) = j + 8*((j>>3)+grp)
// A fragments: AF[((rt*16 + kb)*32 + lane)*16 + j] = Acat[rt*16+m][kb*32+K]
// B fragments: BF[((kb*64 + ct)*32 + lane)*16 + j] = Bcat[kb*32+K][ct*16+n]
// Each fragment is 32 contiguous bytes per lane -> two global_load_b128.

// scatter one value into fragment-ready AF; inverse of the K(j,grp) map:
//   given ko = k&31: grp = (ko>>3)&1, j = (ko&7) + 8*(ko>>4)
__device__ __forceinline__ void scatter_af(_Float16* __restrict__ AF,
                                           int n, int k, float v) {
  const int rt = n >> 4, m = n & 15;
  const int kb = k >> 5, ko = k & 31;
  const int grp = (ko >> 3) & 1;
  const int j   = (ko & 7) + 8 * (ko >> 4);
  const int lane = m + 16 * grp;
  AF[(((rt * 16 + kb) * 32) + lane) * 16 + j] = (_Float16)v;
}

// ---------------------------------------------------------------------------
// K0: one-time init — fragment-ready BF from W_ih/W_hh, biasSum,
//     zero c state, and the t=0 A matrix: [relu(x0@W_in+b_in) | h0=0].
// ---------------------------------------------------------------------------
__global__ __launch_bounds__(256) void init_kernel(
    const float* __restrict__ W_ih, const float* __restrict__ W_hh,
    const float* __restrict__ b_ih, const float* __restrict__ b_hh,
    const float* __restrict__ x, const float* __restrict__ W_in,
    const float* __restrict__ b_in,
    _Float16* __restrict__ BF, _Float16* __restrict__ AF,
    float* __restrict__ biasSum, float* __restrict__ cbuf) {
  const int id = blockIdx.x * 256 + threadIdx.x;
  if (id < KC * G4) {
    const int j    = id & 15;
    const int lane = (id >> 4) & 31;
    const int rest = id >> 9;
    const int ct   = rest & 63;
    const int kb   = rest >> 6;          // 0..15
    const int n    = lane & 15;
    const int grp  = lane >> 4;
    const int K    = j + 8 * ((j >> 3) + grp);
    const int k    = kb * 32 + K;        // 0..511
    const int col  = ct * 16 + n;        // gate column 0..1023
    const float v  = (k < HH) ? W_ih[col * HH + k] : W_hh[col * HH + (k - HH)];
    BF[id] = (_Float16)v;
  }
  if (id < G4) biasSum[id] = b_ih[id] + b_hh[id];
  if (id < NB * HH) {
    cbuf[id] = 0.f;
    const int n = id >> 8, h = id & 255;
    float acc = b_in[h];
#pragma unroll
    for (int f = 0; f < FF; ++f)
      acc = fmaf(x[(size_t)n * LL * FF + f], W_in[f * HH + h], acc);
    scatter_af(AF, n, h, fmaxf(acc, 0.f));   // h_in(t=0)
    scatter_af(AF, n, HH + h, 0.f);          // h_prev = 0
  }
}

// ---------------------------------------------------------------------------
// K2 (per step): gates[1024x1024] = Acat @ Bcat + biasSum, all WMMA.
// No LDS, no barriers: fragments come straight from global (hot in L2).
// WG = 256 thr = 8 waves; each wave computes a 64x32 tile = 4x2 subtiles
// -> per K-block: 6 fragment loads feeding 8 v_wmma_f32_16x16x32_f16.
// ---------------------------------------------------------------------------
__global__ __launch_bounds__(256) void lstm_gemm_kernel(
    const _Float16* __restrict__ AF, const _Float16* __restrict__ BF,
    const float* __restrict__ biasSum, float* __restrict__ gates) {
  const int tid  = threadIdx.x;
  const int lane = tid & 31;
  const int wv   = tid >> 5;
  const int w    = blockIdx.x * 8 + wv;   // 512 waves total
  const int wrow = w >> 5;                // 0..15  -> 64-row strip
  const int wcol = w & 31;                // 0..31  -> 32-col strip

  const v16h* AFv = (const v16h*)AF;      // one fragment per (slot, lane)
  const v16h* BFv = (const v16h*)BF;

  v8f acc[4][2] = {};

#pragma unroll 4
  for (int kb = 0; kb < KC / 32; ++kb) {
    v16h a[4], b[2];
#pragma unroll
    for (int i = 0; i < 4; ++i)
      a[i] = AFv[((wrow * 4 + i) * 16 + kb) * 32 + lane];
#pragma unroll
    for (int j = 0; j < 2; ++j)
      b[j] = BFv[(kb * 64 + wcol * 2 + j) * 32 + lane];
#pragma unroll
    for (int i = 0; i < 4; ++i)
#pragma unroll
      for (int j = 0; j < 2; ++j)
        acc[i][j] = __builtin_amdgcn_wmma_f32_16x16x32_f16(
            false, a[i], false, b[j], (short)0, acc[i][j], false, false);
  }

  const int nn  = lane & 15;
  const int grp = lane >> 4;
  const int row0 = wrow * 64;
  const int col0 = wcol * 32;
  float bs[2];
#pragma unroll
  for (int j = 0; j < 2; ++j) bs[j] = biasSum[col0 + j * 16 + nn];

#pragma unroll
  for (int i = 0; i < 4; ++i)
#pragma unroll
    for (int j = 0; j < 2; ++j)
#pragma unroll
      for (int r = 0; r < 8; ++r) {
        const int mm = r + 8 * grp;       // C/D layout: VGPR r, lane group
        gates[(row0 + i * 16 + mm) * G4 + col0 + j * 16 + nn] =
            acc[i][j][r] + bs[j];
      }
}

// ---------------------------------------------------------------------------
// K3 (per step): LSTM elementwise update + fused fcOut (+sigmoid+scaling)
// + preparation of the NEXT step's A matrix:
//   - scatter h_t (f16 fragments, K cols 256..511)
//   - compute h_in_{t+1} = relu(x[:,t+1,:]@W_in+b_in), scatter (K cols 0..255)
// one block per basin n; thread h handles hidden unit h.
// ---------------------------------------------------------------------------
__global__ __launch_bounds__(256) void lstm_update_kernel(
    const float* __restrict__ gates, float* __restrict__ cbuf,
    const float* __restrict__ W_out, const float* __restrict__ b_out,
    const float* __restrict__ x, const float* __restrict__ W_in,
    const float* __restrict__ b_in, _Float16* __restrict__ AF,
    float* __restrict__ params, int t) {
  __shared__ float xs[FF];
  __shared__ float sh[HH];
  __shared__ float part[OUTP][8];
  const int n = blockIdx.x;
  const int h = threadIdx.x;

  const bool more = (t + 1) < LL;
  if (h < FF && more) xs[h] = x[((size_t)n * LL + t + 1) * FF + h];

  const float gi = gates[n * G4 + h];
  const float gf = gates[n * G4 + HH + h];
  const float gg = gates[n * G4 + 2 * HH + h];
  const float go = gates[n * G4 + 3 * HH + h];
  float c = cbuf[n * HH + h];
  const float si = 1.f / (1.f + expf(-gi));
  const float sf = 1.f / (1.f + expf(-gf));
  const float so = 1.f / (1.f + expf(-go));
  c = sf * c + si * tanhf(gg);
  const float hn = so * tanhf(c);
  cbuf[n * HH + h] = c;
  sh[h] = hn;
  scatter_af(AF, n, HH + h, hn);            // h_t -> next step's A (K 256..511)
  __syncthreads();

  if (more) {                               // fcIn for step t+1
    float acc = b_in[h];
#pragma unroll
    for (int f = 0; f < FF; ++f) acc = fmaf(xs[f], W_in[f * HH + h], acc);
    scatter_af(AF, n, h, fmaxf(acc, 0.f));  // K cols 0..255
  }

  if (h < OUTP * 8) {                       // 8 partial lanes per output
    const int o = h >> 3, p8 = h & 7;
    float s = 0.f;
#pragma unroll
    for (int k = 0; k < 32; ++k) {
      const int hh = p8 * 32 + k;
      s = fmaf(sh[hh], W_out[hh * OUTP + o], s);
    }
    part[o][p8] = s;
  }
  __syncthreads();
  if (h < OUTP) {
    float s = b_out[h];
#pragma unroll
    for (int j = 0; j < 8; ++j) s += part[h][j];
    const float p = 1.f / (1.f + expf(-s));
    params[((size_t)t * NB + n) * OUTP + h] = c_off[h] + c_scale[h] * p;
  }
}

// ---------------------------------------------------------------------------
// K5: ExpHYDRO snow+soil bucket recurrence. thread-per-basin, state in regs.
// ---------------------------------------------------------------------------
__global__ __launch_bounds__(256) void bucket_kernel(
    const float* __restrict__ P, const float* __restrict__ T,
    const float* __restrict__ PET, const float* __restrict__ params,
    float* __restrict__ Q) {
  const int n = blockIdx.x * blockDim.x + threadIdx.x;
  if (n >= NB) return;
  float Sw = 0.001f, Ssl = 0.001f, Sss = 0.001f;
  float a0 = 0.f, a1 = 0.f, a2 = 0.f, a3 = 0.f, a4 = 0.f;  // antecedent T
  for (int t = 0; t < LL; ++t) {
    const float* p = &params[((size_t)t * NB + n) * OUTP];
    const float tmin = p[0], ddf = p[1], tmx = p[2], smax = p[3];
    const float fpar = p[4], qmax = p[5], alpha = p[6], beta = p[7];
    const float Pt = P[(size_t)n * LL + t];
    const float Tt = T[(size_t)n * LL + t];
    const float PEt = PET[(size_t)n * LL + t];
    // snow bucket
    const float pr = (Tt >= tmin) ? Pt : 0.f;
    const float ps = (Tt < tmin) ? Pt : 0.f;
    const float melt = fminf(fmaxf(ddf * (Tt - tmx), 0.f), Sw);
    Sw = Sw - melt + ps;
    a0 = a1; a1 = a2; a2 = a3; a3 = a4; a4 = Tt;
    // soil bucket
    const float Ss = fminf(Sss + Ssl, smax - EPSV);
    const bool freeze = (a0 + a1 + a2 + a3 + a4) * 0.2f < 0.f;
    const float minS = freeze ? Sss : 0.f;
    const float maxS = freeze ? Ss : Sss;
    Sss = minS + (maxS - minS) * alpha;
    Ssl = fmaxf(Ss - Sss, EPSV);
    const float Qb = fminf(fmaxf(qmax * expf(-fpar * (smax - Sss - Ssl)), 0.f),
                           fmaxf(Ssl - EPSV, qmax));
    const float cr = fminf(fmaxf(powf(Ssl / (smax - Sss), beta), 0.f), 1.f);
    const float et = fminf(PEt * cr, Ssl - Qb);
    Ssl = Ssl + pr + melt - et - Qb;
    const float Qs = fmaxf(Ssl - (smax - Sss), 0.f);
    Ssl -= Qs;
    Q[(size_t)n * LL + t] = Qb + Qs;
  }
}

// ---------------------------------------------------------------------------
// K6: gamma unit hydrograph + causal conv. thread-per-basin.
// ---------------------------------------------------------------------------
__global__ __launch_bounds__(256) void route_kernel(
    const float* __restrict__ params, const float* __restrict__ Q,
    float* __restrict__ out) {
  const int n = blockIdx.x * blockDim.x + threadIdx.x;
  if (n >= NB) return;
  float w[TMAXUH];
  float s = 0.f;
#pragma unroll
  for (int k = 0; k < TMAXUH; ++k) {
    const float aa = fmaxf(params[((size_t)k * NB + n) * OUTP + 8], 0.f) + 0.1f;
    const float th = fmaxf(params[((size_t)k * NB + n) * OUTP + 9], 0.f) + 0.5f;
    const float tv = (float)k + 0.5f;
    const float denom = expf(lgammaf(aa)) * powf(th, aa);
    const float wk = powf(tv, aa - 1.f) * expf(-tv / th) / denom;
    w[k] = wk; s += wk;
  }
  const float inv = 1.f / s;
#pragma unroll
  for (int k = 0; k < TMAXUH; ++k) w[k] *= inv;
  for (int l = 0; l < LL; ++l) {
    float acc = 0.f;
    const int km = (l < TMAXUH - 1) ? l : (TMAXUH - 1);
    for (int k = 0; k <= km; ++k) acc = fmaf(w[k], Q[(size_t)n * LL + l - k], acc);
    out[(size_t)n * LL + l] = acc;
  }
}

// ---------------------------------------------------------------------------
extern "C" void kernel_launch(void* const* d_in, const int* in_sizes, int n_in,
                              void* d_out, int out_size, void* d_ws, size_t ws_size,
                              hipStream_t stream) {
  (void)in_sizes; (void)n_in; (void)out_size; (void)ws_size;
  const float* x     = (const float*)d_in[0];
  const float* P     = (const float*)d_in[1];
  const float* T     = (const float*)d_in[2];
  const float* PET   = (const float*)d_in[3];
  const float* W_in  = (const float*)d_in[4];
  const float* b_in  = (const float*)d_in[5];
  const float* W_ih  = (const float*)d_in[6];
  const float* W_hh  = (const float*)d_in[7];
  const float* b_ih  = (const float*)d_in[8];
  const float* b_hh  = (const float*)d_in[9];
  const float* W_out = (const float*)d_in[10];
  const float* b_out = (const float*)d_in[11];
  float* out = (float*)d_out;

  char* ws = (char*)d_ws;
  _Float16* BF     = (_Float16*)(ws);                 // 512*1024*2   = 1,048,576
  float*    biasSum= (float*)   (ws + 1048576);       // 1024*4       = 4,096
  float*    cbuf   = (float*)   (ws + 1052672);       // 1024*256*4   = 1,048,576
  _Float16* AF     = (_Float16*)(ws + 2101248);       // 1024*512*2   = 1,048,576
  float*    gates  = (float*)   (ws + 3149824);       // 1024*1024*4  = 4,194,304
  float*    params = (float*)   (ws + 7344128);       // 1095*1024*10*4 = 44,851,200
  float*    Qbuf   = (float*)   (ws + 52195328);      // 1024*1095*4  = 4,485,120
  // total ~56.7 MB

  // one-time init (weights, biases, c0, and the t=0 A matrix)
  init_kernel<<<(KC * G4 + 255) / 256, 256, 0, stream>>>(
      W_ih, W_hh, b_ih, b_hh, x, W_in, b_in, BF, AF, biasSum, cbuf);

  // recurrent scan: WMMA gates GEMM -> fused update/fcOut/next-A
  for (int t = 0; t < LL; ++t) {
    lstm_gemm_kernel<<<64, 256, 0, stream>>>(AF, BF, biasSum, gates);
    lstm_update_kernel<<<NB, 256, 0, stream>>>(
        gates, cbuf, W_out, b_out, x, W_in, b_in, AF, params, t);
  }

  // bucket recurrence + routing
  bucket_kernel<<<NB / 256, 256, 0, stream>>>(P, T, PET, params, Qbuf);
  route_kernel<<<NB / 256, 256, 0, stream>>>(params, Qbuf, out);
}
